// FasterRCNN_46196668236501
// MI455X (gfx1250) — matrix operand order repaired and checked by reading, not compiled, not run on hardware
//
#include <hip/hip_runtime.h>
#include <hip/hip_bf16.h>

// ---------------------------------------------------------------------------
// Faster-RCNN forward for gfx1250 (MI455X).
// GEMMs use V_WMMA_F32_16X16X32_BF16 (fp32 accumulate, on-the-fly f32->bf16).
// FC GEMM A-tiles staged via the Tensor Data Mover (tensor_load_to_lds) with
// LDS padding producing the [64][17]-dword swizzle directly.
// ---------------------------------------------------------------------------

typedef __attribute__((ext_vector_type(16))) __bf16 v16bf;
typedef __attribute__((ext_vector_type(8)))  float  v8f;
typedef __attribute__((ext_vector_type(4)))  unsigned uint4v;
typedef __attribute__((ext_vector_type(8)))  int      int8v;
typedef __attribute__((ext_vector_type(4)))  int      int4v;

#if defined(__has_builtin)
# if __has_builtin(__builtin_amdgcn_tensor_load_to_lds) && \
     __has_builtin(__builtin_amdgcn_s_wait_tensorcnt)
#  define USE_TDM 1
# endif
#endif
#ifndef USE_TDM
# define USE_TDM 0
#endif

#define DEV static __device__ __forceinline__

DEV int imin(int a, int b) { return a < b ? a : b; }
DEV int imax(int a, int b) { return a > b ? a : b; }

DEV unsigned short f2bf(float f) {
    unsigned u = __builtin_bit_cast(unsigned, f);
    u += 0x7FFFu + ((u >> 16) & 1u);          // round-to-nearest-even
    return (unsigned short)(u >> 16);
}
DEV float bf2f(unsigned short h) {
    unsigned u = ((unsigned)h) << 16;
    return __builtin_bit_cast(float, u);
}
DEV unsigned packbf(float lo, float hi) {
    return (unsigned)f2bf(lo) | ((unsigned)f2bf(hi) << 16);
}

union FragU { unsigned u[8]; v16bf v; };

// A fragment (16x32 bf16): lane m = lane&15, half = lane>=16.
// pair index p(v) = v + (v>=4?4:0) + half*4 ; LDS row stride 17 u32.
DEV void loadA(const unsigned* aB, int mrow, int half, FragU& f) {
    const unsigned* r = aB + mrow * 17 + half * 4;
    f.u[0] = r[0];  f.u[1] = r[1];  f.u[2] = r[2];  f.u[3] = r[3];
    f.u[4] = r[8];  f.u[5] = r[9];  f.u[6] = r[10]; f.u[7] = r[11];
}
// B fragment (32x16 bf16): n = lane&15 (+tile offset), pair row = v + (v>=4?4:0) + half*4.
// LDS: 16 pair-rows x 66 u32 stride.
DEV void loadB(const unsigned* bB, int ncol, int half, FragU& f) {
    const unsigned* c = bB + ncol + half * 4 * 66;
#pragma unroll
    for (int v = 0; v < 8; v++) f.u[v] = c[(v + (v >= 4 ? 4 : 0)) * 66];
}
DEV v8f wmma_bf16(const FragU& a, const FragU& b, v8f c) {
    return __builtin_amdgcn_wmma_f32_16x16x32_bf16(false, a.v, false, b.v,
                                                   (short)0, c, false, false);
}

// ---------------- output layout (f32 elements) ------------------------------
#define OUT_RPN_REG  0          // (2,9216,4)
#define OUT_RPN_CLS  73728      // (2,9216)
#define OUT_NMS_REG  92160      // (2,64,4)
#define OUT_NMS_CLS  92672      // (2,64)
#define OUT_RCNN_REG 92800      // (128,4)
#define OUT_RCNN_CLS 93312      // (128,21)
#define OUT_ANCHORS  96000      // (9216,4)

__device__ __constant__ float c_AH[9] = {32.f,64.f,128.f,32.f,64.f,128.f,64.f,128.f,256.f};
__device__ __constant__ float c_AW[9] = {32.f,64.f,128.f,64.f,128.f,256.f,32.f,64.f,128.f};

DEV float anchor_val(int flat) {  // flat in [0,36864) per image
    int row = flat >> 2, col = flat & 3;
    int a = row >> 10, yy = (row >> 5) & 31, xx = row & 31;
    float cy = (yy + 0.5f) * 16.0f, cx = (xx + 0.5f) * 16.0f;
    float ah = c_AH[a], aw = c_AW[a];
    return (col == 0) ? cy - 0.5f * ah
         : (col == 1) ? cx - 0.5f * aw
         : (col == 2) ? cy + 0.5f * ah
                      : cx + 0.5f * aw;
}

// ---------------------------------------------------------------------------
// K0: anchors output
__global__ void k_anchors(float* __restrict__ out) {
    int g = blockIdx.x * blockDim.x + threadIdx.x;
    if (g >= 36864) return;
    out[OUT_ANCHORS + g] = anchor_val(g);
}

// ---------------------------------------------------------------------------
// K1: stem patchify conv as GEMM.  M=2048 (b,y,x)  N=1024 (Cout)  K=768.
// Writes fm f32 NCHW (for ROI pool) and fm bf16 NHWC (for 3x3 conv A-tiles).
__global__ __launch_bounds__(256) void k_stem(
    const float* __restrict__ img, const float* __restrict__ wcnn,
    const float* __restrict__ bcnn, float* __restrict__ fm32,
    unsigned short* __restrict__ fmb) {
    __shared__ unsigned aP[64][17];
    __shared__ unsigned bP[16][66];
    int tid = threadIdx.x, lane = tid & 31, wave = tid >> 5;
    int mblk = wave & 3, nhalf = wave >> 2, half = (lane >> 4) & 1;
    int Mt = blockIdx.x * 64, Nt = blockIdx.y * 64;
    v8f c0 = {}, c1 = {};

    for (int k0 = 0; k0 < 768; k0 += 32) {
        __syncthreads();
        for (int s = tid; s < 1024; s += 256) {           // A tile 64m x 16 pairs
            int ml = s >> 4, p = s & 15;
            int m = Mt + ml;
            int bb = m >> 10, hw = m & 1023, y = hw >> 5, x = hw & 31;
            int k = k0 + 2 * p;
            int cc = k >> 8, r = k & 255, kh = r >> 4, kw = r & 15;
            const float* src = img + ((size_t)(bb * 3 + cc) * 512 + (y * 16 + kh)) * 512
                                   + (x * 16 + kw);
            aP[ml][p] = packbf(src[0], src[1]);
        }
        for (int s = tid; s < 1024; s += 256) {           // B tile 16 pairs x 64n
            int n = s >> 4, p = s & 15;
            const float* src = wcnn + (size_t)(Nt + n) * 768 + k0 + 2 * p;
            bP[p][n] = packbf(src[0], src[1]);
        }
        __syncthreads();
        FragU a, b0, b1;
        loadA(&aP[0][0], mblk * 16 + (lane & 15), half, a);
        loadB(&bP[0][0], (nhalf * 2 + 0) * 16 + (lane & 15), half, b0);
        loadB(&bP[0][0], (nhalf * 2 + 1) * 16 + (lane & 15), half, b1);
        c0 = wmma_bf16(a, b0, c0);
        c1 = wmma_bf16(a, b1, c1);
    }
#pragma unroll
    for (int t = 0; t < 2; t++) {
        v8f acc = t ? c1 : c0;
        int nblk = nhalf * 2 + t;
#pragma unroll
        for (int v = 0; v < 8; v++) {
            int m = Mt + mblk * 16 + v + (half ? 8 : 0);
            int n = Nt + nblk * 16 + (lane & 15);
            int bb = m >> 10, hw = m & 1023, y = hw >> 5, x = hw & 31;
            float val = fmaxf(acc[v] + bcnn[n], 0.0f);
            fm32[((size_t)(bb * 1024 + n) * 32 + y) * 32 + x] = val;
            fmb[((size_t)(bb * 32 + y) * 32 + x) * 1024 + n] = f2bf(val);
        }
    }
}

// ---------------------------------------------------------------------------
// K2: RPN 3x3 conv (pad 1) as GEMM.  M=2048  N=1024  K=9216 = 9 taps x 1024c.
__global__ __launch_bounds__(256) void k_rpn3x3(
    const unsigned short* __restrict__ fmb, const float* __restrict__ wrpn,
    const float* __restrict__ brpn, unsigned short* __restrict__ hb) {
    __shared__ unsigned aP[64][17];
    __shared__ unsigned bP[16][66];
    int tid = threadIdx.x, lane = tid & 31, wave = tid >> 5;
    int mblk = wave & 3, nhalf = wave >> 2, half = (lane >> 4) & 1;
    int Mt = blockIdx.x * 64, Nt = blockIdx.y * 64;
    v8f c0 = {}, c1 = {};

    for (int pos = 0; pos < 9; pos++) {
        int dy = pos / 3 - 1, dx = pos % 3 - 1;
        for (int ck = 0; ck < 1024; ck += 32) {
            __syncthreads();
            for (int s = tid; s < 1024; s += 256) {       // A tile (zero-padded halo)
                int ml = s >> 4, p = s & 15;
                int m = Mt + ml;
                int bb = m >> 10, hw = m & 1023, y = hw >> 5, x = hw & 31;
                int yy = y + dy, xx = x + dx;
                unsigned uval = 0u;
                if (yy >= 0 && yy < 32 && xx >= 0 && xx < 32)
                    uval = *(const unsigned*)(fmb + ((size_t)(bb * 32 + yy) * 32 + xx) * 1024
                                                   + ck + 2 * p);
                aP[ml][p] = uval;
            }
            for (int s = tid; s < 1024; s += 256) {       // B tile, k = c*9+pos
                int n = s >> 4, p = s & 15;
                const float* w = wrpn + (size_t)(Nt + n) * 9216 + (ck + 2 * p) * 9 + pos;
                bP[p][n] = packbf(w[0], w[9]);
            }
            __syncthreads();
            FragU a, b0, b1;
            loadA(&aP[0][0], mblk * 16 + (lane & 15), half, a);
            loadB(&bP[0][0], (nhalf * 2 + 0) * 16 + (lane & 15), half, b0);
            loadB(&bP[0][0], (nhalf * 2 + 1) * 16 + (lane & 15), half, b1);
            c0 = wmma_bf16(a, b0, c0);
            c1 = wmma_bf16(a, b1, c1);
        }
    }
#pragma unroll
    for (int t = 0; t < 2; t++) {
        v8f acc = t ? c1 : c0;
        int nblk = nhalf * 2 + t;
#pragma unroll
        for (int v = 0; v < 8; v++) {
            int m = Mt + mblk * 16 + v + (half ? 8 : 0);
            int n = Nt + nblk * 16 + (lane & 15);
            int bb = m >> 10, hw = m & 1023;
            float val = fmaxf(acc[v] + brpn[n], 0.0f);
            hb[((size_t)bb * 1024 + hw) * 1024 + n] = f2bf(val);
        }
    }
}

// ---------------------------------------------------------------------------
// K3: 1x1 RPN heads (reg: 36ch, cls: 9ch) + rpn_reg_abs (= reg + anchors).
__global__ void k_rpnheads(
    const unsigned short* __restrict__ hb, const float* __restrict__ wreg,
    const float* __restrict__ breg, const float* __restrict__ wcls,
    const float* __restrict__ bcls, float* __restrict__ out,
    float* __restrict__ regabs) {
    int g = blockIdx.x * blockDim.x + threadIdx.x;
    if (g >= 2 * 45 * 1024) return;
    int b = g / (45 * 1024);
    int rem = g - b * 45 * 1024;
    int co = rem >> 10, hw = rem & 1023;
    const unsigned short* hp = hb + ((size_t)b * 1024 + hw) * 1024;
    const float* w;
    float bias;
    if (co < 36) { w = wreg + co * 1024;        bias = breg[co]; }
    else         { w = wcls + (co - 36) * 1024; bias = bcls[co - 36]; }
    float acc = 0.0f;
    for (int c = 0; c < 1024; c += 4) {
        acc += bf2f(hp[c + 0]) * w[c + 0] + bf2f(hp[c + 1]) * w[c + 1]
             + bf2f(hp[c + 2]) * w[c + 2] + bf2f(hp[c + 3]) * w[c + 3];
    }
    acc += bias;
    if (co < 36) {
        int flat = co * 1024 + hw;
        out[OUT_RPN_REG + b * 36864 + flat] = acc;
        regabs[b * 36864 + flat] = acc + anchor_val(flat);
    } else {
        out[OUT_RPN_CLS + b * 9216 + (co - 36) * 1024 + hw] = acc;
    }
}

// ---------------------------------------------------------------------------
// K4: per-image NMS (64 picks) + stable top-64 order + ROI quantization.
// One 1024-thread workgroup per image; 9216-entry work array in LDS.
__global__ __launch_bounds__(1024) void k_nms(
    float* __restrict__ out, const float* __restrict__ regabs,
    int* __restrict__ roisI) {
    __shared__ float s_work[9216];
    __shared__ float s_rv[1024];
    __shared__ int   s_ri[1024];
    __shared__ int   s_picks[64];
    __shared__ int   s_valid[64];
    __shared__ int   s_order[64];
    __shared__ float s_b[5];
    __shared__ int   s_nv;

    int tid = threadIdx.x;
    int b = blockIdx.x;
    const float* scores = out + OUT_RPN_CLS + b * 9216;
    const float* boxes  = regabs + b * 36864;

    for (int i = tid; i < 9216; i += 1024) {
        float s = scores[i];
        s_work[i] = 1.0f / (1.0f + expf(-s));   // sigmoid
    }
    __syncthreads();

    // ---- NMS loop: argmax + IoU suppression, 64 rounds ----
    for (int it = 0; it < 64; it++) {
        float bv = -3.0e38f; int bi = 0;
        for (int i = tid; i < 9216; i += 1024) {
            float v = s_work[i];
            if (v > bv) { bv = v; bi = i; }
        }
        s_rv[tid] = bv; s_ri[tid] = bi;
        __syncthreads();
        for (int off = 512; off > 0; off >>= 1) {
            if (tid < off) {
                float v2 = s_rv[tid + off]; int i2 = s_ri[tid + off];
                if (v2 > s_rv[tid] || (v2 == s_rv[tid] && i2 < s_ri[tid])) {
                    s_rv[tid] = v2; s_ri[tid] = i2;
                }
            }
            __syncthreads();
        }
        if (tid == 0) {
            int j = s_ri[0];
            s_picks[it] = j;
            s_valid[it] = (s_rv[0] > -0.5f) ? 1 : 0;
            float y1 = boxes[j*4], x1 = boxes[j*4+1], y2 = boxes[j*4+2], x2 = boxes[j*4+3];
            s_b[0] = y1; s_b[1] = x1; s_b[2] = y2; s_b[3] = x2;
            s_b[4] = (y2 - y1) * (x2 - x1);
        }
        __syncthreads();
        float by1 = s_b[0], bx1 = s_b[1], by2 = s_b[2], bx2 = s_b[3], aj = s_b[4];
        for (int i = tid; i < 9216; i += 1024) {
            float y1 = boxes[i*4], x1 = boxes[i*4+1], y2 = boxes[i*4+2], x2 = boxes[i*4+3];
            float yy1 = fmaxf(by1, y1), xx1 = fmaxf(bx1, x1);
            float yy2 = fminf(by2, y2), xx2 = fminf(bx2, x2);
            float inter = fmaxf(yy2 - yy1, 0.0f) * fmaxf(xx2 - xx1, 0.0f);
            float ai = (y2 - y1) * (x2 - x1);
            float iou = inter / (aj + ai - inter);
            if (iou > 0.7f) s_work[i] = -1.0f;
        }
        __syncthreads();
    }

    // ---- stable top-64 of raw scores (argsort(-scores)[0:64]) ----
    for (int i = tid; i < 9216; i += 1024) s_work[i] = scores[i];
    __syncthreads();
    for (int it = 0; it < 64; it++) {
        float bv = -3.4e38f; int bi = 0;
        for (int i = tid; i < 9216; i += 1024) {
            float v = s_work[i];
            if (v > bv) { bv = v; bi = i; }
        }
        s_rv[tid] = bv; s_ri[tid] = bi;
        __syncthreads();
        for (int off = 512; off > 0; off >>= 1) {
            if (tid < off) {
                float v2 = s_rv[tid + off]; int i2 = s_ri[tid + off];
                if (v2 > s_rv[tid] || (v2 == s_rv[tid] && i2 < s_ri[tid])) {
                    s_rv[tid] = v2; s_ri[tid] = i2;
                }
            }
            __syncthreads();
        }
        if (tid == 0) { int j = s_ri[0]; s_order[it] = j; s_work[j] = -3.4e38f; }
        __syncthreads();
    }

    if (tid == 0) {
        int nv = 0;
        for (int i = 0; i < 64; i++) nv += s_valid[i];
        s_nv = nv;
    }
    __syncthreads();

    if (tid < 64) {
        int i = tid;
        int sel;
        if (s_valid[i]) sel = s_picks[i];
        else {
            int q = i - s_nv;
            q = imax(q, 0); q = imin(q, 63);
            sel = s_order[q];
        }
        float y1 = boxes[sel*4], x1 = boxes[sel*4+1], y2 = boxes[sel*4+2], x2 = boxes[sel*4+3];
        int base = OUT_NMS_REG + (b * 64 + i) * 4;
        out[base + 0] = y1; out[base + 1] = x1; out[base + 2] = y2; out[base + 3] = x2;
        out[OUT_NMS_CLS + b * 64 + i] = scores[sel];
        int n = b * 64 + i;   // ROI quantization: floor(x*S), ceil(x*S) with S=1/16
        roisI[n * 4 + 0] = (int)floorf(x1 * 0.0625f);
        roisI[n * 4 + 1] = (int)floorf(y1 * 0.0625f);
        roisI[n * 4 + 2] = (int)ceilf(x2 * 0.0625f);
        roisI[n * 4 + 3] = (int)ceilf(y2 * 0.0625f);
    }
}

// ---------------------------------------------------------------------------
// K5: ROI max-pool 7x7 -> x (128, 50176) stored bf16 (FC GEMM A-matrix order
// k = c*49 + ph*7 + pw, matching rois.reshape(128,-1)).
__global__ __launch_bounds__(256) void k_roipool(
    const float* __restrict__ fm32, const int* __restrict__ roisI,
    unsigned short* __restrict__ xb) {
    int n = blockIdx.x;            // roi 0..127
    int b = n >> 6;
    int xs = roisI[n*4+0], ys = roisI[n*4+1], xe = roisI[n*4+2], ye = roisI[n*4+3];
    float rw = (float)imax(xe - xs + 1, 1);
    float rh = (float)imax(ye - ys + 1, 1);
    int hs[7], he[7], wss[7], wee[7];
#pragma unroll
    for (int p = 0; p < 7; p++) {
        int a0 = (int)floorf((float)p * rh / 7.0f) + ys;
        int a1 = (int)ceilf((float)(p + 1) * rh / 7.0f) + ys;
        hs[p] = imin(imax(a0, 0), 32);
        he[p] = imin(imax(a1, 0), 32);
        a0 = (int)floorf((float)p * rw / 7.0f) + xs;
        a1 = (int)ceilf((float)(p + 1) * rw / 7.0f) + xs;
        wss[p] = imin(imax(a0, 0), 32);
        wee[p] = imin(imax(a1, 0), 32);
    }
    for (int c = threadIdx.x; c < 1024; c += 256) {
        const float* f = fm32 + (size_t)(b * 1024 + c) * 1024;
        unsigned short* o = xb + (size_t)n * 50176 + c * 49;
        for (int ph = 0; ph < 7; ph++) {
            for (int pw = 0; pw < 7; pw++) {
                float m = -3.4e38f;
                bool any = false;
                for (int y = hs[ph]; y < he[ph]; y++)
                    for (int x = wss[pw]; x < wee[pw]; x++) {
                        m = fmaxf(m, f[y * 32 + x]);
                        any = true;
                    }
                o[ph * 7 + pw] = f2bf(any ? m : 0.0f);
            }
        }
    }
}

// ---------------------------------------------------------------------------
// K6: FC GEMM, split-K.  M=128  N=1024  K=50176, 16-way K split so the 205MB
// w_fc read parallelizes across WGPs (bandwidth-bound stage).
// A tile (pure bf16 copy) staged by the Tensor Data Mover: the D# uses LDS
// padding (1 dword per 16 dwords) to produce the [64][17] layout directly.
__global__ __launch_bounds__(256) void k_fc(
    const unsigned short* __restrict__ xb, const float* __restrict__ wfc,
    float* __restrict__ fcpart) {
    __shared__ unsigned aP[64][17];
    __shared__ unsigned bP[16][66];
    int tid = threadIdx.x, lane = tid & 31, wave = tid >> 5;
    int mblk = wave & 3, nhalf = wave >> 2, half = (lane >> 4) & 1;
    int Mt = blockIdx.x * 64, Nt = blockIdx.y * 64;
    int ks = blockIdx.z;
    int kbeg = ks * 3136, kend = kbeg + 3136;
    v8f c0 = {}, c1 = {};
#if USE_TDM
    unsigned lds_a = (unsigned)(uintptr_t)(&aP[0][0]);   // LDS byte offset
#endif

    for (int k0 = kbeg; k0 < kend; k0 += 32) {
        __syncthreads();
#if USE_TDM
        if (wave == 0) {
            // 2-D TDM copy: 16 dwords x 64 rows, row stride 25088 dwords,
            // pad 1 dword after every 16 dwords -> aP[64][17].
            unsigned long long ga = (unsigned long long)(uintptr_t)xb
                + ((unsigned long long)Mt * 50176ull + (unsigned long long)k0) * 2ull;
            uint4v g0 = { 1u,                                   // count=1 (valid D#)
                          lds_a,                                // lds_addr
                          (unsigned)(ga & 0xFFFFFFFFull),       // global_addr lo
                          (unsigned)((ga >> 32) & 0x01FFFFFFull) | (2u << 30) }; // hi|type=2
            int8v g1 = { (int)((2u << 16) | (1u << 20) | (3u << 22)), // 4B elems, pad_en, 16dw interval, 1dw pad
                         (int)(16u << 16),                      // tensor_dim0 = 16
                         (int)(64u << 16),                      // tensor_dim1 = 64
                         (int)(16u << 16),                      // tile_dim0 = 16
                         64,                                    // tile_dim1 = 64
                         25088,                                 // tensor_dim0_stride (dwords)
                         0, 0 };
            int4v gz = { 0, 0, 0, 0 };
#if defined(__clang_major__) && __clang_major__ >= 23
            int8v gz8 = { 0, 0, 0, 0, 0, 0, 0, 0 };
            __builtin_amdgcn_tensor_load_to_lds(g0, g1, gz, gz, gz8, 0);
#else
            __builtin_amdgcn_tensor_load_to_lds(g0, g1, gz, gz, 0);
#endif
        }
#else
        for (int s = tid; s < 1024; s += 256) {           // A tile (contiguous in k)
            int ml = s >> 4, p = s & 15;
            aP[ml][p] = *(const unsigned*)(xb + (size_t)(Mt + ml) * 50176 + k0 + 2 * p);
        }
#endif
        for (int s = tid; s < 1024; s += 256) {           // B tile (w_fc is k-major)
            int nn = s & 63, p = s >> 6;
            const float* w = wfc + (size_t)(k0 + 2 * p) * 1024 + Nt + nn;
            __builtin_prefetch(w + 32 * 1024, 0, 0);      // next K tile -> gfx1250 prefetch
            bP[p][nn] = packbf(w[0], w[1024]);
        }
#if USE_TDM
        if (wave == 0) __builtin_amdgcn_s_wait_tensorcnt(0);
#endif
        __syncthreads();
        FragU a, b0, b1;
        loadA(&aP[0][0], mblk * 16 + (lane & 15), half, a);
        loadB(&bP[0][0], (nhalf * 2 + 0) * 16 + (lane & 15), half, b0);
        loadB(&bP[0][0], (nhalf * 2 + 1) * 16 + (lane & 15), half, b1);
        c0 = wmma_bf16(a, b0, c0);
        c1 = wmma_bf16(a, b1, c1);
    }
#pragma unroll
    for (int t = 0; t < 2; t++) {
        v8f acc = t ? c1 : c0;
        int nblk = nhalf * 2 + t;
#pragma unroll
        for (int v = 0; v < 8; v++) {
            int m = Mt + mblk * 16 + v + (half ? 8 : 0);
            int n = Nt + nblk * 16 + (lane & 15);
            fcpart[(size_t)ks * 131072 + (size_t)m * 1024 + n] = acc[v];
        }
    }
}

// K7: reduce split-K partials + bias + relu -> hfc (128,1024)
__global__ void k_fcreduce(const float* __restrict__ fcpart,
                           const float* __restrict__ bfc,
                           float* __restrict__ hfc) {
    int g = blockIdx.x * blockDim.x + threadIdx.x;
    if (g >= 128 * 1024) return;
    int n = g & 1023;
    float acc = bfc[n];
#pragma unroll
    for (int s = 0; s < 16; s++) acc += fcpart[(size_t)s * 131072 + g];
    hfc[g] = fmaxf(acc, 0.0f);
}

// K8: tiny rcnn heads (128x4 and 128x21)
__global__ void k_heads(const float* __restrict__ hfc,
                        const float* __restrict__ wrreg, const float* __restrict__ brreg,
                        const float* __restrict__ wrcls, const float* __restrict__ brcls,
                        float* __restrict__ out) {
    int g = blockIdx.x * blockDim.x + threadIdx.x;
    if (g >= 128 * 25) return;
    int m = g / 25, j = g % 25;
    const float* h = hfc + (size_t)m * 1024;
    if (j < 4) {
        float acc = 0.0f;
        for (int c = 0; c < 1024; c++) acc += h[c] * wrreg[c * 4 + j];
        out[OUT_RCNN_REG + m * 4 + j] = acc + brreg[j];
    } else {
        int jj = j - 4;
        float acc = 0.0f;
        for (int c = 0; c < 1024; c++) acc += h[c] * wrcls[c * 21 + jj];
        out[OUT_RCNN_CLS + m * 21 + jj] = acc + brcls[jj];
    }
}

// ---------------------------------------------------------------------------
extern "C" void kernel_launch(void* const* d_in, const int* in_sizes, int n_in,
                              void* d_out, int out_size, void* d_ws, size_t ws_size,
                              hipStream_t stream) {
    (void)in_sizes; (void)n_in; (void)out_size; (void)ws_size;
    const float* img   = (const float*)d_in[0];
    const float* wcnn  = (const float*)d_in[1];
    const float* bcnn  = (const float*)d_in[2];
    const float* wrpn  = (const float*)d_in[3];
    const float* brpn  = (const float*)d_in[4];
    const float* wreg  = (const float*)d_in[5];
    const float* breg  = (const float*)d_in[6];
    const float* wcls  = (const float*)d_in[7];
    const float* bcls  = (const float*)d_in[8];
    const float* wfc   = (const float*)d_in[9];
    const float* bfc   = (const float*)d_in[10];
    const float* wrreg = (const float*)d_in[11];
    const float* brreg = (const float*)d_in[12];
    const float* wrcls = (const float*)d_in[13];
    const float* brcls = (const float*)d_in[14];
    float* out = (float*)d_out;

    char* ws = (char*)d_ws;
    float*          fm32   = (float*)(ws + 0);                 // 2*1024*32*32 f32
    unsigned short* fmb    = (unsigned short*)(ws + 8388608);  // NHWC bf16
    unsigned short* hb     = (unsigned short*)(ws + 12582912); // NHWC bf16
    float*          regabs = (float*)(ws + 16777216);          // (2,36864)
    int*            roisI  = (int*)(ws + 17072128);            // (128,4)
    unsigned short* xb     = (unsigned short*)(ws + 17074176); // (128,50176) bf16
    float*          fcpart = (float*)(ws + 29919232);          // (16,128,1024)
    float*          hfc    = (float*)(ws + 38307840);          // (128,1024)

    k_anchors <<<dim3(144),       256, 0, stream>>>(out);
    k_stem    <<<dim3(32, 16),    256, 0, stream>>>(img, wcnn, bcnn, fm32, fmb);
    k_rpn3x3  <<<dim3(32, 16),    256, 0, stream>>>(fmb, wrpn, brpn, hb);
    k_rpnheads<<<dim3(360),       256, 0, stream>>>(hb, wreg, breg, wcls, bcls, out, regabs);
    k_nms     <<<dim3(2),        1024, 0, stream>>>(out, regabs, roisI);
    k_roipool <<<dim3(128),       256, 0, stream>>>(fm32, roisI, xb);
    k_fc      <<<dim3(2, 16, 16), 256, 0, stream>>>(xb, wfc, fcpart);
    k_fcreduce<<<dim3(512),       256, 0, stream>>>(fcpart, bfc, hfc);
    k_heads   <<<dim3(13),        256, 0, stream>>>(hfc, wrreg, brreg, wrcls, brcls, out);
}
